// SlotAttentionGroupingVQ_58076547776860
// MI455X (gfx1250) — matrix operand order repaired and checked by reading, not compiled, not run on hardware
//
#include <hip/hip_runtime.h>

// ---------------- types ----------------
typedef __attribute__((ext_vector_type(16))) __bf16 bv16;
typedef __attribute__((ext_vector_type(8)))  __bf16 bv8;
typedef __attribute__((ext_vector_type(8)))  float  fv8;

__device__ __forceinline__ __bf16 f2bf(float f) {
  unsigned u = __builtin_bit_cast(unsigned, f);
  u += 0x7fffu + ((u >> 16) & 1u);
  unsigned short s = (unsigned short)(u >> 16);
  return __builtin_bit_cast(__bf16, s);
}

// ---------------- problem constants ----------------
#define BSZ 32
#define NF  1024
#define FDM 768
#define NSL 24
#define DD  256
#define KVQ 256
#define CCB 512
#define HH  4
#define DH  64

// ============================================================
// Wave32 WMMA GEMM, 16x64 output per wave (A fragment reused 4x).
//   C[m,n] = sum_k A[m,k] * B(k,n) (+bias[n])
//   A: bf16 row-major (lda);  B: bf16 "weight layout" B(k,n)=Bm[n*ldb+k]
//   CT=0: C[m,n] at cof + m*ldc + n         (f32 or bf16 via OUTBF)
//   CT=1: transposed scatter for V: off = (m/cMB)*sCb + n*ldc + (m%cMB)
//   Batched over grid.z = b*H2 + h with separate b/h strides (elements).
// ============================================================
template<bool CT, bool OUTBF, bool BIAS>
__global__ void gemm_wmma(const __bf16* __restrict__ A, long long sAb, long long sAh, int lda,
                          const __bf16* __restrict__ Bm, long long sBb, long long sBh, int ldb,
                          void* __restrict__ Cv, long long sCb, long long sCh, int ldc, int cMB,
                          const float* __restrict__ bias, int H2,
                          int M, int N, int K) {
  const int lane = threadIdx.x;
  const int z  = blockIdx.z;
  const int bb = z / H2, hh = z % H2;
  A  += (long long)bb * sAb + (long long)hh * sAh;
  Bm += (long long)bb * sBb + (long long)hh * sBh;
  const int m0 = blockIdx.x * 16;
  const int n0 = blockIdx.y * 64;
  const int l15    = lane & 15;
  const int hiHalf = lane >> 4;          // 0 or 1
  const int rowA = m0 + l15;
  const int kbA  = hiHalf * 8;           // A lane K-base (ISA 16-bit A layout)
  const int kbB  = hiHalf * 8;           // B lane K-base (ISA 16-bit B layout)
  fv8 acc[4] = {{}, {}, {}, {}};
  const __bf16* bp0 = Bm + (long long)(n0 + l15) * ldb + kbB;
  for (int k0 = 0; k0 < K; k0 += 32) {
    bv16 a = {};
    if (rowA < M) {
      const __bf16* ap = A + (long long)rowA * lda + k0 + kbA;
      bv8 lo = *(const bv8*)(ap);        // K = k0+kbA .. +7
      bv8 hi = *(const bv8*)(ap + 16);   // K = k0+kbA+16 .. +23
      a = __builtin_shufflevector(lo, hi, 0,1,2,3,4,5,6,7,8,9,10,11,12,13,14,15);
      if (k0 + 32 < K) __builtin_prefetch((const void*)(ap + 32), 0, 1);
    }
#pragma unroll
    for (int t = 0; t < 4; ++t) {
      const __bf16* bp = bp0 + (long long)(t * 16) * ldb + k0;
      bv8 lo = *(const bv8*)(bp);
      bv8 hi = *(const bv8*)(bp + 16);
      bv16 b = __builtin_shufflevector(lo, hi, 0,1,2,3,4,5,6,7,8,9,10,11,12,13,14,15);
      acc[t] = __builtin_amdgcn_wmma_f32_16x16x32_bf16(false, a, false, b, (short)0,
                                                       acc[t], false, false);
    }
  }
  const int mbase = m0 + hiHalf * 8;
  const long long cof = (long long)bb * sCb + (long long)hh * sCh;
#pragma unroll
  for (int t = 0; t < 4; ++t) {
    const int n = n0 + t * 16 + l15;
    const float bs = BIAS ? bias[n] : 0.f;
#pragma unroll
    for (int r = 0; r < 8; ++r) {
      int m = mbase + r;
      if (m < M) {
        float v = acc[t][r] + bs;
        long long off;
        if (CT) off = (long long)(m / cMB) * sCb + (long long)n * ldc + (m % cMB);
        else    off = cof + (long long)m * ldc + n;
        if (OUTBF) ((__bf16*)Cv)[off] = f2bf(v);
        else       ((float*)Cv)[off] = v;
      }
    }
  }
}

// ---------------- LayerNorm over last dim, f32 in -> bf16 out ----------------
__global__ void ln_kernel(const float* __restrict__ x, const float* __restrict__ g,
                          const float* __restrict__ b, __bf16* __restrict__ y, int cols) {
  const int row = blockIdx.x;
  const float* xr = x + (long long)row * cols;
  __shared__ float s1[256], s2[256];
  float s = 0.f, ss = 0.f;
  for (int c = threadIdx.x; c < cols; c += 256) { float v = xr[c]; s += v; ss += v * v; }
  s1[threadIdx.x] = s; s2[threadIdx.x] = ss; __syncthreads();
  for (int o = 128; o > 0; o >>= 1) {
    if (threadIdx.x < o) { s1[threadIdx.x] += s1[threadIdx.x + o]; s2[threadIdx.x] += s2[threadIdx.x + o]; }
    __syncthreads();
  }
  float m   = s1[0] / (float)cols;
  float var = s2[0] / (float)cols - m * m;
  float inv = rsqrtf(var + 1e-5f);
  for (int c = threadIdx.x; c < cols; c += 256)
    y[(long long)row * cols + c] = f2bf((xr[c] - m) * inv * g[c] + b[c]);
}

// ---------------- f32 -> bf16 convert ----------------
__global__ void conv_bf16(const float* __restrict__ x, __bf16* __restrict__ y, long long n) {
  long long t = (long long)blockIdx.x * blockDim.x + threadIdx.x;
  if (t < n) y[t] = f2bf(x[t]);
}

// -------- softmax over slot axis: attn[bh][i][j], softmax over i --------
__global__ void softmax_slots(float* __restrict__ attn, int BH, int NS, int N, float scale) {
  long long t = (long long)blockIdx.x * blockDim.x + threadIdx.x;
  if (t >= (long long)BH * N) return;
  int bh = (int)(t / N), j = (int)(t % N);
  float* p = attn + ((long long)bh * NS) * N + j;
  float mx = -1e30f;
  for (int i = 0; i < NS; ++i) mx = fmaxf(mx, p[(long long)i * N] * scale);
  float s = 0.f;
  for (int i = 0; i < NS; ++i) {
    float e = __expf(p[(long long)i * N] * scale - mx);
    p[(long long)i * N] = e; s += e;
  }
  float inv = 1.f / s;
  for (int i = 0; i < NS; ++i) p[(long long)i * N] *= inv;
}

// -------- weighted-mean normalization over features: w = (attn+eps)/rowsum --------
__global__ void wnorm_kernel(const float* __restrict__ attn, __bf16* __restrict__ w, int N) {
  const int row = blockIdx.x;              // (b*H+h)*NS + i
  const float* a = attn + (long long)row * N;
  __shared__ float sh[256];
  float s = 0.f;
  for (int j = threadIdx.x; j < N; j += 256) s += a[j] + 1e-8f;
  sh[threadIdx.x] = s; __syncthreads();
  for (int o = 128; o > 0; o >>= 1) { if (threadIdx.x < o) sh[threadIdx.x] += sh[threadIdx.x + o]; __syncthreads(); }
  float inv = 1.f / sh[0];
  for (int j = threadIdx.x; j < N; j += 256)
    w[(long long)row * N + j] = f2bf((a[j] + 1e-8f) * inv);
}

// -------- attn_vis[b,i,j] = mean over heads --------
__global__ void attn_vis_kernel(const float* __restrict__ attn, float* __restrict__ vis,
                                int B, int H, int NS, int N) {
  long long t = (long long)blockIdx.x * blockDim.x + threadIdx.x;
  if (t >= (long long)B * NS * N) return;
  int j = (int)(t % N);
  long long r = t / N;
  int i = (int)(r % NS), b = (int)(r / NS);
  float s = 0.f;
  for (int h = 0; h < H; ++h) s += attn[((((long long)b * H + h) * NS) + i) * N + j];
  vis[t] = s * (1.f / (float)H);
}

// -------- GRU combine --------
__global__ void gru_combine(const float* __restrict__ gi, const float* __restrict__ gh,
                            const float* __restrict__ h, float* __restrict__ out,
                            int rows, int D) {
  long long t = (long long)blockIdx.x * blockDim.x + threadIdx.x;
  if (t >= (long long)rows * D) return;
  int row = (int)(t / D), d = (int)(t % D);
  const float* gir = gi + (long long)row * 3 * D;
  const float* ghr = gh + (long long)row * 3 * D;
  float r = 1.f / (1.f + __expf(-(gir[d] + ghr[d])));
  float z = 1.f / (1.f + __expf(-(gir[D + d] + ghr[D + d])));
  float n = tanhf(gir[2 * D + d] + r * ghr[2 * D + d]);
  out[t] = (1.f - z) * n + z * h[t];
}

// -------- codebook row sum-of-squares --------
__global__ void rowsumsq(const float* __restrict__ cb, float* __restrict__ cc, int D) {
  const int c = blockIdx.x;
  __shared__ float sh[256];
  float s = 0.f;
  for (int d = threadIdx.x; d < D; d += 256) { float v = cb[(long long)c * D + d]; s += v * v; }
  sh[threadIdx.x] = s; __syncthreads();
  for (int o = 128; o > 0; o >>= 1) { if (threadIdx.x < o) sh[threadIdx.x] += sh[threadIdx.x + o]; __syncthreads(); }
  if (threadIdx.x == 0) cc[c] = sh[0];
}

// -------- VQ: argmin over codebook, gather, per-row commit partial --------
__global__ void vq_argmin(const float* __restrict__ S, const float* __restrict__ cc,
                          const float* __restrict__ x, const float* __restrict__ cb,
                          float* __restrict__ qslots, float* __restrict__ idx_out,
                          float* __restrict__ rowsq, int C, int D) {
  const int row = blockIdx.x;
  __shared__ float sv[256];
  __shared__ int   si[256];
  float best = 3.4e38f; int bi = 0;
  for (int c = threadIdx.x; c < C; c += 256) {
    float v = cc[c] - 2.f * S[(long long)row * C + c];
    if (v < best) { best = v; bi = c; }
  }
  sv[threadIdx.x] = best; si[threadIdx.x] = bi; __syncthreads();
  for (int o = 128; o > 0; o >>= 1) {
    if (threadIdx.x < o) {
      float ov = sv[threadIdx.x + o]; int oi = si[threadIdx.x + o];
      if (ov < sv[threadIdx.x] || (ov == sv[threadIdx.x] && oi < si[threadIdx.x])) {
        sv[threadIdx.x] = ov; si[threadIdx.x] = oi;
      }
    }
    __syncthreads();
  }
  const int idx = si[0];
  __syncthreads();
  float part = 0.f;
  for (int d = threadIdx.x; d < D; d += 256) {
    float q = cb[(long long)idx * D + d];
    qslots[(long long)row * D + d] = q;
    float df = x[(long long)row * D + d] - q;
    part += df * df;
  }
  sv[threadIdx.x] = part; __syncthreads();
  for (int o = 128; o > 0; o >>= 1) { if (threadIdx.x < o) sv[threadIdx.x] += sv[threadIdx.x + o]; __syncthreads(); }
  if (threadIdx.x == 0) { rowsq[row] = sv[0]; idx_out[row] = (float)idx; }
}

__global__ void vq_finalize(const float* __restrict__ rowsq, float* __restrict__ out,
                            int rows, float invtotal) {
  __shared__ float sh[256];
  float s = 0.f;
  for (int r = threadIdx.x; r < rows; r += 256) s += rowsq[r];
  sh[threadIdx.x] = s; __syncthreads();
  for (int o = 128; o > 0; o >>= 1) { if (threadIdx.x < o) sh[threadIdx.x] += sh[threadIdx.x + o]; __syncthreads(); }
  if (threadIdx.x == 0) *out = sh[0] * invtotal;
}

// ============================================================
// Workspace layout (bytes, all 256-aligned)
// ============================================================
static const size_t OFF_LNF   = 0;                        // 32768*768 bf16
static const size_t OFF_K     = OFF_LNF   + 50331648;     // K  [B,N,KVQ] bf16
static const size_t OFF_VT    = OFF_K     + 16777216;     // V^T [B,H,dh,N] bf16
static const size_t OFF_SL    = OFF_VT    + 16777216;     // slots f32 768*256
static const size_t OFF_SLB   = OFF_SL    + 786432;       // slots bf16
static const size_t OFF_LNS   = OFF_SLB   + 393216;       // LN(slots) bf16
static const size_t OFF_Q     = OFF_LNS   + 393216;       // q bf16 768*256
static const size_t OFF_ATTN  = OFF_Q     + 393216;       // attn f32 32*4*24*1024
static const size_t OFF_W     = OFF_ATTN  + 12582912;     // w bf16 32*4*24*1024
static const size_t OFF_UPD   = OFF_W     + 6291456;      // upd bf16 768*256
static const size_t OFF_GI    = OFF_UPD   + 393216;       // gi f32 768*768
static const size_t OFF_GH    = OFF_GI    + 2359296;      // gh f32 768*768
static const size_t OFF_WQ    = OFF_GH    + 2359296;      // bf16 weights
static const size_t OFF_WK    = OFF_WQ    + 131072;
static const size_t OFF_WV    = OFF_WK    + 393216;
static const size_t OFF_WIH   = OFF_WV    + 393216;
static const size_t OFF_WHH   = OFF_WIH   + 393216;
static const size_t OFF_CB    = OFF_WHH   + 393216;       // codebook bf16 512*256
static const size_t OFF_CC    = OFF_CB    + 262144;       // cc f32 512
static const size_t OFF_S     = OFF_CC    + 2048;         // scores f32 768*512
static const size_t OFF_QS    = OFF_S     + 1572864;      // q_slots f32 768*256
static const size_t OFF_RSQ   = OFF_QS    + 786432;       // per-row commit sums

// d_out layout (floats): slots | post_attn | pre_attn | indices | commit
static const size_t OUT_SLOTS = 0;                        // 196608
static const size_t OUT_POST  = 196608;                   // 786432
static const size_t OUT_PRE   = OUT_POST + 786432;        // 786432
static const size_t OUT_IDX   = OUT_PRE  + 786432;        // 768
static const size_t OUT_COMMIT= OUT_IDX  + 768;           // 1

extern "C" void kernel_launch(void* const* d_in, const int* in_sizes, int n_in,
                              void* d_out, int out_size, void* d_ws, size_t ws_size,
                              hipStream_t stream) {
  (void)in_sizes; (void)n_in; (void)out_size; (void)ws_size;
  const float* features     = (const float*)d_in[0];
  const float* conditioning = (const float*)d_in[1];
  const float* Wq   = (const float*)d_in[2];
  const float* Wk   = (const float*)d_in[3];
  const float* Wv   = (const float*)d_in[4];
  const float* g_in = (const float*)d_in[5];
  const float* b_in = (const float*)d_in[6];
  const float* g_s  = (const float*)d_in[7];
  const float* b_s  = (const float*)d_in[8];
  const float* Wih  = (const float*)d_in[9];
  const float* Whh  = (const float*)d_in[10];
  const float* bih  = (const float*)d_in[11];
  const float* bhh  = (const float*)d_in[12];
  const float* codebook = (const float*)d_in[13];

  char* ws = (char*)d_ws;
  __bf16* LNF = (__bf16*)(ws + OFF_LNF);
  __bf16* KB  = (__bf16*)(ws + OFF_K);
  __bf16* VT  = (__bf16*)(ws + OFF_VT);
  float*  SL  = (float*)(ws + OFF_SL);
  __bf16* SLB = (__bf16*)(ws + OFF_SLB);
  __bf16* LNS = (__bf16*)(ws + OFF_LNS);
  __bf16* QB  = (__bf16*)(ws + OFF_Q);
  float*  AT  = (float*)(ws + OFF_ATTN);
  __bf16* WB  = (__bf16*)(ws + OFF_W);
  __bf16* UPD = (__bf16*)(ws + OFF_UPD);
  float*  GIb = (float*)(ws + OFF_GI);
  float*  GHb = (float*)(ws + OFF_GH);
  __bf16* WQb = (__bf16*)(ws + OFF_WQ);
  __bf16* WKb = (__bf16*)(ws + OFF_WK);
  __bf16* WVb = (__bf16*)(ws + OFF_WV);
  __bf16* WIHb= (__bf16*)(ws + OFF_WIH);
  __bf16* WHHb= (__bf16*)(ws + OFF_WHH);
  __bf16* CBb = (__bf16*)(ws + OFF_CB);
  float*  CCp = (float*)(ws + OFF_CC);
  float*  SS  = (float*)(ws + OFF_S);
  float*  QS  = (float*)(ws + OFF_QS);
  float*  RSQ = (float*)(ws + OFF_RSQ);
  float*  out = (float*)d_out;

  const dim3 W32(32, 1, 1);

  // ---- Phase 0: weight conversion + codebook norms ----
  conv_bf16<<<(65536 + 255) / 256, 256, 0, stream>>>(Wq, WQb, 65536);
  conv_bf16<<<(196608 + 255) / 256, 256, 0, stream>>>(Wk, WKb, 196608);
  conv_bf16<<<(196608 + 255) / 256, 256, 0, stream>>>(Wv, WVb, 196608);
  conv_bf16<<<(196608 + 255) / 256, 256, 0, stream>>>(Wih, WIHb, 196608);
  conv_bf16<<<(196608 + 255) / 256, 256, 0, stream>>>(Whh, WHHb, 196608);
  conv_bf16<<<(131072 + 255) / 256, 256, 0, stream>>>(codebook, CBb, 131072);
  rowsumsq<<<CCB, 256, 0, stream>>>(codebook, CCp, DD);

  // ---- Phase 1: LN(features), K, V (slot-independent -> computed ONCE) ----
  ln_kernel<<<BSZ * NF, 256, 0, stream>>>(features, g_in, b_in, LNF, FDM);
  // K: standard layout [B*N, KVQ]
  gemm_wmma<false, true, false><<<dim3(2048, 4, 1), W32, 0, stream>>>(
      LNF, 0, 0, FDM, WKb, 0, 0, FDM, KB, 0, 0, KVQ, 0, nullptr, 1, BSZ * NF, KVQ, FDM);
  // V: scattered directly into transposed layout [B, H*dh, N] (per b: n*1024 + j)
  gemm_wmma<true, true, false><<<dim3(2048, 4, 1), W32, 0, stream>>>(
      LNF, 0, 0, FDM, WVb, 0, 0, FDM, VT, (long long)KVQ * NF, 0, NF, NF,
      nullptr, 1, BSZ * NF, KVQ, FDM);

  // slots = conditioning
  hipMemcpyAsync(SL, conditioning, (size_t)BSZ * NSL * DD * sizeof(float),
                 hipMemcpyDeviceToDevice, stream);

  for (int call = 0; call < 2; ++call) {
    float* vis_out = out + (call == 0 ? OUT_PRE : OUT_POST);
    for (int it = 0; it < 3; ++it) {
      const float* s_in = (call == 1 && it == 0) ? QS : SL;
      float* s_out = (call == 1 && it == 2) ? (out + OUT_SLOTS) : SL;

      ln_kernel<<<BSZ * NSL, 256, 0, stream>>>(s_in, g_s, b_s, LNS, DD);
      conv_bf16<<<(768 * 256 + 255) / 256, 256, 0, stream>>>(s_in, SLB, 768 * 256);

      // q = LN(slots) @ Wq^T   [768,256]
      gemm_wmma<false, true, false><<<dim3(48, 4, 1), W32, 0, stream>>>(
          LNS, 0, 0, DD, WQb, 0, 0, DD, QB, 0, 0, KVQ, 0, nullptr, 1, 768, KVQ, DD);

      // dots[b,h,i,j] = q . k  per (b,h):  [24,64] x [64,1024]; B = K rows (BT)
      gemm_wmma<false, false, false><<<dim3(2, 16, BSZ * HH), W32, 0, stream>>>(
          QB, (long long)NSL * KVQ, DH, KVQ,
          KB, (long long)NF * KVQ, DH, KVQ,
          AT, (long long)HH * NSL * NF, (long long)NSL * NF, NF, 0,
          nullptr, HH, NSL, NF, DH);

      // softmax over slots (axis i), scale = dh^-0.5
      softmax_slots<<<(BSZ * HH * NF + 255) / 256, 256, 0, stream>>>(
          AT, BSZ * HH, NSL, NF, 0.125f);

      if (it == 2)
        attn_vis_kernel<<<(BSZ * NSL * NF + 255) / 256, 256, 0, stream>>>(
            AT, vis_out, BSZ, HH, NSL, NF);

      // w = (attn + eps) / rowsum_j  -> bf16
      wnorm_kernel<<<BSZ * HH * NSL, 256, 0, stream>>>(AT, WB, NF);

      // upd[b,i,h,d] = sum_j w[b,h,i,j] v[b,j,h,d]:  [24,1024] x [1024,64]
      // B = V^T rows (n=d contiguous in k=j): base (b*256 + h*64)*1024
      gemm_wmma<false, true, false><<<dim3(2, 1, BSZ * HH), W32, 0, stream>>>(
          WB, (long long)HH * NSL * NF, (long long)NSL * NF, NF,
          VT, (long long)KVQ * NF, (long long)DH * NF, NF,
          UPD, (long long)NSL * KVQ, DH, KVQ, 0,
          nullptr, HH, NSL, DH, NF);

      // GRU gates: gi = upd @ Wih^T + bih ; gh = h @ Whh^T + bhh   [768,768]
      gemm_wmma<false, false, true><<<dim3(48, 12, 1), W32, 0, stream>>>(
          UPD, 0, 0, KVQ, WIHb, 0, 0, KVQ, GIb, 0, 0, 3 * DD, 0, bih, 1, 768, 3 * DD, KVQ);
      gemm_wmma<false, false, true><<<dim3(48, 12, 1), W32, 0, stream>>>(
          SLB, 0, 0, DD, WHHb, 0, 0, DD, GHb, 0, 0, 3 * DD, 0, bhh, 1, 768, 3 * DD, DD);

      gru_combine<<<(768 * DD + 255) / 256, 256, 0, stream>>>(GIb, GHb, s_in, s_out, 768, DD);
    }

    if (call == 0) {
      // ---- VQ pooling on slots ----
      conv_bf16<<<(768 * 256 + 255) / 256, 256, 0, stream>>>(SL, SLB, 768 * 256);
      gemm_wmma<false, false, false><<<dim3(48, 8, 1), W32, 0, stream>>>(
          SLB, 0, 0, DD, CBb, 0, 0, DD, SS, 0, 0, CCB, 0, nullptr, 1, 768, CCB, DD);
      vq_argmin<<<768, 256, 0, stream>>>(SS, CCp, SL, codebook, QS, out + OUT_IDX, RSQ, CCB, DD);
      vq_finalize<<<1, 256, 0, stream>>>(RSQ, out + OUT_COMMIT, 768, 1.f / (768.f * 256.f));
    }
  }
}
